// GATModule_48249662603979
// MI455X (gfx1250) — matrix operand order repaired and checked by reading, not compiled, not run on hardware
//
#include <hip/hip_runtime.h>
#include <hip/hip_fp16.h>

typedef __attribute__((ext_vector_type(16))) _Float16 v16h;
typedef __attribute__((ext_vector_type(8)))  _Float16 v8h;
typedef __attribute__((ext_vector_type(8)))  float    v8f;
typedef __attribute__((ext_vector_type(4)))  float    v4f;

#define BS 8          // B*S graphs
#define NN 1024       // nodes per graph (L)
#define CC 128        // channels
#define IMG 256
#define NEG_SLOPE 0.2f
#define LN_EPS 1e-5f

#define SHUF16(lo, hi) __builtin_shufflevector(lo, hi, 0, 1, 2, 3, 4, 5, 6, 7, \
                                               8, 9, 10, 11, 12, 13, 14, 15)

// ---------------------------------------------------------------------------
// Adjacency as bitmask: adj[(g*NN + i)*32 + (j>>5)] bit (j&31)
// ---------------------------------------------------------------------------
__global__ void k_init_adj(unsigned int* __restrict__ adj) {
  int idx = blockIdx.x * blockDim.x + threadIdx.x;   // BS*NN*32
  if (idx >= BS * NN * 32) return;
  int w   = idx & 31;
  int row = idx >> 5;
  int i   = row & (NN - 1);
  adj[idx] = (w == (i >> 5)) ? (1u << (i & 31)) : 0u;   // self loop
}

__global__ void k_edges(const int* __restrict__ seg, unsigned int* __restrict__ adj) {
  int idx = blockIdx.x * blockDim.x + threadIdx.x;   // BS*IMG*IMG
  if (idx >= BS * IMG * IMG) return;
  int g = idx >> 16;
  int p = idx & 65535;
  int r = p >> 8, c = p & 255;
  const int* s = seg + g * IMG * IMG;
  int a = s[p];
  auto addE = [&](int bA, int bB) {
    if (bA > 0 && bB > 0 && bA != bB) {
      int ai = bA - 1, bi = bB - 1;
      atomicOr(&adj[(g * NN + ai) * 32 + (bi >> 5)], 1u << (bi & 31));
      atomicOr(&adj[(g * NN + bi) * 32 + (ai >> 5)], 1u << (ai & 31));
    }
  };
  if (c < IMG - 1)                addE(a, s[p + 1]);            // horizontal
  if (r < IMG - 1)                addE(a, s[p + IMG]);          // vertical
  if (r < IMG - 1 && c < IMG - 1) addE(a, s[p + IMG + 1]);      // diag '\'
  if (r < IMG - 1 && c < IMG - 1) addE(s[p + 1], s[p + IMG]);   // diag '/'
}

// ---------------------------------------------------------------------------
// Converts. WMMA B-operands are stored SWIZZLED:
//   swz(k, n) = ((k/32)*CC + n)*32 + (k%32)
// so a lane's 16 B halves (k = e + (e&8) + 8*half, fixed n) are two 8-element runs (2xB128).
// ---------------------------------------------------------------------------
__global__ void k_cvt_f32_f16(const float* __restrict__ src, _Float16* __restrict__ dst, int n) {
  int i = blockIdx.x * blockDim.x + threadIdx.x;
  if (i < n) dst[i] = (_Float16)src[i];
}

__global__ void k_cvt_wT_swz(const float* __restrict__ w, _Float16* __restrict__ wsw) {
  int i = blockIdx.x * blockDim.x + threadIdx.x;     // CC*CC
  if (i >= CC * CC) return;
  int o = i / CC, c = i % CC;                        // B[k=c][n=o] = W[o][c]
  wsw[(((c >> 5) * CC) + o) * 32 + (c & 31)] = (_Float16)w[o * CC + c];
}

__global__ void k_cvt_swz(const float* __restrict__ wf, _Float16* __restrict__ o) {
  int i = blockIdx.x * blockDim.x + threadIdx.x;     // BS*NN*CC
  if (i >= BS * NN * CC) return;
  int rg = i / CC, n = i % CC;
  int g = rg >> 10, k = rg & (NN - 1);
  o[((size_t)(g * (NN / 32) + (k >> 5)) * CC + n) * 32 + (k & 31)] = (_Float16)wf[i];
}

// ---------------------------------------------------------------------------
// Dense GEMM: Out(f32, M x 128) = A(f16 row-major) @ Bsw(f16 swizzled).
// One wave per 16-row strip x all 128 cols, 8 f32 WMMA accumulators.
// A: 2x b128 per K-step; B: 2x b128 per tile (L2-hot, 32KB total).
// grid.x = M/128, block = 256.
// ---------------------------------------------------------------------------
__global__ void k_gemm_f16(const _Float16* __restrict__ A, const _Float16* __restrict__ Bsw,
                           float* __restrict__ Out) {
  int tid = threadIdx.x;
  int wave = tid >> 5, lane = tid & 31;
  int m = lane & 15, half = lane >> 4;
  int tM = blockIdx.x * 8 + wave;
  int row = tM * 16 + m;
  int n0 = lane & 15;
  v8f acc[8] = {};
#pragma unroll
  for (int kk = 0; kk < CC; kk += 32) {
    v8h a0 = *(const v8h*)&A[row * CC + kk + 8 * half];
    v8h a1 = *(const v8h*)&A[row * CC + kk + 16 + 8 * half];
    v16h a = SHUF16(a0, a1);
#pragma unroll
    for (int nt = 0; nt < 8; ++nt) {
      const _Float16* bp = &Bsw[(((kk >> 5) * CC) + nt * 16 + n0) * 32 + 8 * half];
      v16h b = SHUF16(*(const v8h*)bp, *(const v8h*)(bp + 16));
      acc[nt] = __builtin_amdgcn_wmma_f32_16x16x32_f16(false, a, false, b, (short)0,
                                                       acc[nt], false, false);
    }
  }
#pragma unroll
  for (int nt = 0; nt < 8; ++nt)
#pragma unroll
    for (int v = 0; v < 8; ++v) {
      int ro = tM * 16 + v + 8 * half;
      Out[ro * CC + nt * 16 + n0] = acc[nt][v];
    }
}

// ---------------------------------------------------------------------------
// Per-node attention scores: es/ed[rg*heads + h] = <Wf[rg, h*d : h*d+d], a_src/dst[h]>
// ---------------------------------------------------------------------------
__global__ void k_scores(const float* __restrict__ Wf, const float* __restrict__ aSrc,
                         const float* __restrict__ aDst, float* __restrict__ es,
                         float* __restrict__ ed, int heads, int d) {
  int idx = blockIdx.x * blockDim.x + threadIdx.x;   // BS*NN*heads
  if (idx >= BS * NN * heads) return;
  int h  = idx % heads;
  int rg = idx / heads;
  const float* wrow = Wf + rg * CC + h * d;
  float s = 0.f, t = 0.f;
  for (int k = 0; k < d; ++k) {
    float v = wrow[k];
    s += v * aSrc[h * d + k];
    t += v * aDst[h * d + k];
  }
  es[idx] = s;
  ed[idx] = t;
}

// ---------------------------------------------------------------------------
// Row max / sum for the masked softmax. Block = 8 waves, all same (g, head);
// ed column async-staged into LDS. Each wave handles 8 rows.
// grid.x = BS * HEADS * 16, block = 256.
// ---------------------------------------------------------------------------
template <int HEADS>
__global__ void k_row_stats(const unsigned int* __restrict__ adj,
                            const float* __restrict__ es, const float* __restrict__ ed,
                            float* __restrict__ rmax, float* __restrict__ rsum) {
  __shared__ float sED[NN];
  int bid  = blockIdx.x;
  int g    = bid / (HEADS * 16);
  int rem  = bid % (HEADS * 16);
  int head = rem / 16;
  int blk  = rem % 16;
  int tid  = threadIdx.x;
#pragma unroll
  for (int t = 0; t < NN / 256; ++t) {
    int j = t * 256 + tid;
    unsigned dsa = (unsigned)(uintptr_t)&sED[j];
    const float* gp = ed + (size_t)((g << 10) + j) * HEADS + head;
    asm volatile("global_load_async_to_lds_b32 %0, %1, off"
                 :: "v"(dsa), "v"(gp) : "memory");
  }
  asm volatile("s_wait_asynccnt 0x0" ::: "memory");
  __syncthreads();

  int wave = tid >> 5, lane = tid & 31;
  for (int rr = 0; rr < 8; ++rr) {
    int i  = blk * 64 + wave * 8 + rr;
    int rg = (g << 10) + i;
    float esv = es[rg * HEADS + head];
    const unsigned int* arow = adj + rg * 32;
    float m = -3.4e38f;
    float ev[32];
#pragma unroll
    for (int t = 0; t < 32; ++t) {
      unsigned int w = arow[t];
      float e = esv + sED[t * 32 + lane];
      e = (e >= 0.f) ? e : NEG_SLOPE * e;
      bool bit = (w >> lane) & 1u;
      ev[t] = bit ? e : -3.4e38f;
      if (bit) m = fmaxf(m, e);
    }
    for (int off = 16; off; off >>= 1) m = fmaxf(m, __shfl_xor(m, off, 32));
    float s = 0.f;
#pragma unroll
    for (int t = 0; t < 32; ++t)
      if (ev[t] > -3.0e38f) s += __expf(ev[t] - m);
    for (int off = 16; off; off >>= 1) s += __shfl_xor(s, off, 32);
    if (lane == 0) { rmax[rg * HEADS + head] = m; rsum[rg * HEADS + head] = s; }
  }
}

// ---------------------------------------------------------------------------
// Fused masked-softmax x Wf contraction via WMMA.
// Block = 8 waves sharing (g, head); ed async-staged into LDS.
// Wave builds P tile once per K-step, feeds D/16 accumulators; B tiles are
// 2x global_load_b128 from the swizzled f16 Wf. grid.x = BS*HEADS*8, block=256.
// ---------------------------------------------------------------------------
template <int HEADS, int D>
__global__ void k_attn_wmma(const unsigned int* __restrict__ adj,
                            const float* __restrict__ es, const float* __restrict__ ed,
                            const float* __restrict__ rmax, const float* __restrict__ rsum,
                            const _Float16* __restrict__ Wfsw,
                            _Float16* __restrict__ outH, float* __restrict__ outF) {
  constexpr int NT = D / 16;
  __shared__ float sED[NN];
  int bid  = blockIdx.x;
  int g    = bid / (HEADS * 8);
  int rem  = bid % (HEADS * 8);
  int head = rem / 8;
  int blkM = rem % 8;
  int tid  = threadIdx.x;
#pragma unroll
  for (int t = 0; t < NN / 256; ++t) {
    int j = t * 256 + tid;
    unsigned dsa = (unsigned)(uintptr_t)&sED[j];
    const float* gp = ed + (size_t)((g << 10) + j) * HEADS + head;
    asm volatile("global_load_async_to_lds_b32 %0, %1, off"
                 :: "v"(dsa), "v"(gp) : "memory");
  }
  asm volatile("s_wait_asynccnt 0x0" ::: "memory");
  __syncthreads();

  int wave = tid >> 5, lane = tid & 31;
  int m = lane & 15, half = lane >> 4;
  int tM = blkM * 8 + wave;
  int i  = tM * 16 + m;
  int rg = (g << 10) + i;
  int rh = rg * HEADS + head;
  float esv  = es[rh];
  float mx   = rmax[rh];
  float rinv = 1.f / rsum[rh];
  int n0 = head * D + (lane & 15);
  const unsigned int* arow = adj + rg * 32;
  const _Float16* wb = Wfsw + (size_t)g * NN * CC;   // per-graph swizzled panel
  v8f acc[NT] = {};
  for (int kk = 0; kk < NN; kk += 32) {
    int kt = kk >> 5;
    if (kt + 1 < NN / 32)
      __builtin_prefetch(wb + ((size_t)(kt + 1) * CC + n0) * 32, 0, 0);
    unsigned int aw = arow[kt];
    // vectorized LDS reads of the 16 ed values this lane needs
    v4f q0 = *(const v4f*)&sED[kk + 8 * half];
    v4f q1 = *(const v4f*)&sED[kk + 8 * half + 4];
    v4f q2 = *(const v4f*)&sED[kk + 16 + 8 * half];
    v4f q3 = *(const v4f*)&sED[kk + 16 + 8 * half + 4];
    float edv[16];
#pragma unroll
    for (int e = 0; e < 4; ++e) {
      edv[e] = q0[e]; edv[4 + e] = q1[e]; edv[8 + e] = q2[e]; edv[12 + e] = q3[e];
    }
    v16h a;
#pragma unroll
    for (int e = 0; e < 16; ++e) {
      int ka = e + (e & 8) + 8 * half;               // A 16x32 k-index
      float evv = esv + edv[e];
      evv = (evv >= 0.f) ? evv : NEG_SLOPE * evv;
      float pj = ((aw >> ka) & 1u) ? __expf(evv - mx) * rinv : 0.f;
      a[e] = (_Float16)pj;
    }
#pragma unroll
    for (int t = 0; t < NT; ++t) {
      const _Float16* bp = wb + ((size_t)kt * CC + n0 + t * 16) * 32 + 8 * half;
      v16h b = SHUF16(*(const v8h*)bp, *(const v8h*)(bp + 16));
      acc[t] = __builtin_amdgcn_wmma_f32_16x16x32_f16(false, a, false, b, (short)0,
                                                      acc[t], false, false);
    }
  }
#pragma unroll
  for (int t = 0; t < NT; ++t)
#pragma unroll
    for (int v = 0; v < 8; ++v) {
      int ro  = (g << 10) + tM * 16 + v + 8 * half;
      int col = n0 + t * 16;
      float x = acc[t][v];
      if (outH) {
        float y = (x > 0.f) ? x : (__expf(x) - 1.f);   // ELU
        outH[ro * CC + col] = (_Float16)y;
      } else {
        outF[ro * CC + col] = x;
      }
    }
}

// ---------------------------------------------------------------------------
// Residual + LayerNorm epilogue. One wave per row (128 ch -> 4 per lane).
// ---------------------------------------------------------------------------
__global__ void k_final(const float* __restrict__ feats, const float* __restrict__ h2,
                        const float* __restrict__ gamma, const float* __restrict__ beta,
                        float* __restrict__ out) {
  int gtid = blockIdx.x * blockDim.x + threadIdx.x;
  int row  = gtid >> 5;
  int lane = gtid & 31;
  if (row >= BS * NN) return;
  float y[4];
  float s = 0.f, sq = 0.f;
#pragma unroll
  for (int t = 0; t < 4; ++t) {
    int c = lane + 32 * t;
    float v = feats[row * CC + c] + h2[row * CC + c];
    y[t] = v; s += v; sq += v * v;
  }
  for (int off = 16; off; off >>= 1) {
    s  += __shfl_xor(s, off, 32);
    sq += __shfl_xor(sq, off, 32);
  }
  float mean = s * (1.f / CC);
  float var  = sq * (1.f / CC) - mean * mean;
  float rstd = rsqrtf(var + LN_EPS);
#pragma unroll
  for (int t = 0; t < 4; ++t) {
    int c = lane + 32 * t;
    out[row * CC + c] = (y[t] - mean) * rstd * gamma[c] + beta[c];
  }
}

// ---------------------------------------------------------------------------
extern "C" void kernel_launch(void* const* d_in, const int* in_sizes, int n_in,
                              void* d_out, int out_size, void* d_ws, size_t ws_size,
                              hipStream_t stream) {
  (void)in_sizes; (void)n_in; (void)out_size; (void)ws_size;
  const float* feats = (const float*)d_in[0];
  const int*   segI  = (const int*)d_in[1];
  // d_in[2] = seg_nums (unused by reference math)
  const float* W1   = (const float*)d_in[3];
  const float* aS1  = (const float*)d_in[4];
  const float* aD1  = (const float*)d_in[5];
  const float* W2   = (const float*)d_in[6];
  const float* aS2  = (const float*)d_in[7];
  const float* aD2  = (const float*)d_in[8];
  const float* gam  = (const float*)d_in[9];
  const float* bet  = (const float*)d_in[10];
  float* out = (float*)d_out;

  char* ws = (char*)d_ws;
  size_t off = 0;
  auto alloc = [&](size_t bytes) -> void* {
    void* p = ws + off;
    off = (off + bytes + 255) & ~(size_t)255;
    return p;
  };
  unsigned int* adj  = (unsigned int*)alloc((size_t)BS * NN * 32 * 4);
  _Float16* Xh    = (_Float16*)alloc((size_t)BS * NN * CC * 2);
  _Float16* W1sw  = (_Float16*)alloc((size_t)CC * CC * 2);
  _Float16* W2sw  = (_Float16*)alloc((size_t)CC * CC * 2);
  float* Wf1  = (float*)alloc((size_t)BS * NN * CC * 4);
  _Float16* Wf1sw = (_Float16*)alloc((size_t)BS * NN * CC * 2);
  float* es1  = (float*)alloc((size_t)BS * NN * 4 * 4);
  float* ed1  = (float*)alloc((size_t)BS * NN * 4 * 4);
  float* mx1  = (float*)alloc((size_t)BS * NN * 4 * 4);
  float* sm1  = (float*)alloc((size_t)BS * NN * 4 * 4);
  _Float16* h1h = (_Float16*)alloc((size_t)BS * NN * CC * 2);
  float* Wf2  = (float*)alloc((size_t)BS * NN * CC * 4);
  _Float16* Wf2sw = (_Float16*)alloc((size_t)BS * NN * CC * 2);
  float* es2  = (float*)alloc((size_t)BS * NN * 4);
  float* ed2  = (float*)alloc((size_t)BS * NN * 4);
  float* mx2  = (float*)alloc((size_t)BS * NN * 4);
  float* sm2  = (float*)alloc((size_t)BS * NN * 4);
  float* h2f  = (float*)alloc((size_t)BS * NN * CC * 4);

  // 1) adjacency bitmask
  k_init_adj<<<(BS * NN * 32 + 255) / 256, 256, 0, stream>>>(adj);
  k_edges<<<(BS * IMG * IMG + 255) / 256, 256, 0, stream>>>(segI, adj);

  // 2) f16 conversions (weights stored pre-swizzled for the WMMA B layout)
  k_cvt_f32_f16<<<(BS * NN * CC + 255) / 256, 256, 0, stream>>>(feats, Xh, BS * NN * CC);
  k_cvt_wT_swz<<<(CC * CC + 255) / 256, 256, 0, stream>>>(W1, W1sw);
  k_cvt_wT_swz<<<(CC * CC + 255) / 256, 256, 0, stream>>>(W2, W2sw);

  const int M = BS * NN;

  // 3) Layer 1
  k_gemm_f16<<<M / 128, 256, 0, stream>>>(Xh, W1sw, Wf1);
  k_cvt_swz<<<(M * CC + 255) / 256, 256, 0, stream>>>(Wf1, Wf1sw);
  k_scores<<<(M * 4 + 255) / 256, 256, 0, stream>>>(Wf1, aS1, aD1, es1, ed1, 4, 32);
  k_row_stats<4><<<BS * 4 * 16, 256, 0, stream>>>(adj, es1, ed1, mx1, sm1);
  k_attn_wmma<4, 32><<<BS * 4 * 8, 256, 0, stream>>>(adj, es1, ed1, mx1, sm1, Wf1sw,
                                                     h1h, nullptr);

  // 4) Layer 2
  k_gemm_f16<<<M / 128, 256, 0, stream>>>(h1h, W2sw, Wf2);
  k_cvt_swz<<<(M * CC + 255) / 256, 256, 0, stream>>>(Wf2, Wf2sw);
  k_scores<<<(M + 255) / 256, 256, 0, stream>>>(Wf2, aS2, aD2, es2, ed2, 1, 128);
  k_row_stats<1><<<BS * 1 * 16, 256, 0, stream>>>(adj, es2, ed2, mx2, sm2);
  k_attn_wmma<1, 128><<<BS * 1 * 8, 256, 0, stream>>>(adj, es2, ed2, mx2, sm2, Wf2sw,
                                                      nullptr, h2f);

  // 5) residual + LayerNorm
  k_final<<<(M * 32 + 255) / 256, 256, 0, stream>>>(feats, h2f, gam, bet, out);
}
